// VQVAE_34608846471822
// MI455X (gfx1250) — matrix operand order repaired and checked by reading, not compiled, not run on hardware
//
#include <hip/hip_runtime.h>

// ---------- problem constants (from reference setup) ----------
#define BATCH   64
#define TT      1024
#define DD      512          // embedding dim
#define KK      1024         // codebook size
#define NN      (BATCH*TT)   // 65536 rows
#define NDTOT   (NN*DD)      // 33554432 quantized elements

// ---------- trivially-copyable vector types ----------
typedef __attribute__((ext_vector_type(4)))  float    f32x4;
typedef __attribute__((ext_vector_type(4)))  unsigned u32x4;
typedef __attribute__((ext_vector_type(8)))  float    v8f;
typedef __attribute__((ext_vector_type(16))) __bf16   v16bf;
typedef __attribute__((ext_vector_type(4)))  __bf16   v4bf;

union ABu { v16bf v; u32x4 q[2]; };   // one WMMA A/B fragment (8 VGPRs)

#define LROW 1040   // padded LDS bytes per bf16 row (1024B data + 16B pad -> 4-bank rotate)
#define SX_BYTES  (128 * LROW)            // 128 input rows, bf16, padded
#define SCB_BYTES (64 * LROW)             // 64 codebook rows, bf16, padded
#define SC2_BYTES (KK * 4)                // all code norms
#define SMEM_BYTES (SX_BYTES + SCB_BYTES + SC2_BYTES)   // 203,776 B < 320 KB

// ---- optional CDNA5 async global->LDS path (ASYNCcnt-tracked) ----
#define AS1 __attribute__((address_space(1)))
#define AS3 __attribute__((address_space(3)))
typedef int v4i_vs __attribute__((vector_size(4 * sizeof(int))));  // builtin's pointee type
#if defined(__has_builtin)
#  if __has_builtin(__builtin_amdgcn_global_load_async_to_lds_b128) && \
      __has_builtin(__builtin_amdgcn_s_wait_asynccnt)
#    define USE_ASYNC_LDS 1
#  endif
#endif
#ifndef USE_ASYNC_LDS
#  define USE_ASYNC_LDS 0
#endif

// =====================================================================
// Kernel 1: codebook prep — per-code squared norm, fp32->bf16 copy,
//           zero histogram + SSE accumulator.  grid=K, block=128.
// =====================================================================
__global__ __launch_bounds__(128)
void vq_prep(const float* __restrict__ cb, float* __restrict__ c2,
             unsigned short* __restrict__ cb16u, unsigned* __restrict__ hist,
             float* __restrict__ sse)
{
    const int k = blockIdx.x;
    const int t = threadIdx.x;                       // 0..127, 4 elems each
    __bf16* cb16 = (__bf16*)cb16u;

    f32x4 f = *(const f32x4*)(cb + (size_t)k * DD + t * 4);
    v4bf o;
    o[0] = (__bf16)f[0]; o[1] = (__bf16)f[1];
    o[2] = (__bf16)f[2]; o[3] = (__bf16)f[3];
    *(v4bf*)(cb16 + (size_t)k * DD + t * 4) = o;

    float s = f[0]*f[0] + f[1]*f[1] + f[2]*f[2] + f[3]*f[3];
    __shared__ float red[128];
    red[t] = s;
    __syncthreads();
    for (int off = 64; off; off >>= 1) {
        if (t < off) red[t] += red[t + off];
        __syncthreads();
    }
    if (t == 0) {
        c2[k]   = red[0];
        hist[k] = 0u;                                 // re-zero every launch
        if (k == 0) sse[0] = 0.0f;
    }
}

// =====================================================================
// Kernel 2: fused distance-GEMM (bf16 WMMA) + argmin + gather +
//           loss partial + histogram.  grid=N/128, block=256 (8 waves).
// =====================================================================
__global__ __launch_bounds__(256)
void vq_main(const float* __restrict__ x, const float* __restrict__ cb,
             const unsigned short* __restrict__ cb16u,
             const float* __restrict__ c2,
             unsigned* __restrict__ hist, float* __restrict__ sse,
             float* __restrict__ out)
{
    extern __shared__ unsigned char smem[];
    unsigned char* sX  = smem;                       // input tile (bf16, padded rows)
    unsigned char* sCB = smem + SX_BYTES;            // codebook tile (bf16, padded rows)
    float*         sC2 = (float*)(smem + SX_BYTES + SCB_BYTES);

    const __bf16* cb16 = (const __bf16*)cb16u;
    const int tid  = threadIdx.x;
    const int lane = tid & 31;
    const int half = lane >> 4;                // 0 or 1 (wave32 halves)
    const int l16  = lane & 15;
    const int wave = tid >> 5;
    const long blockBase = (long)blockIdx.x * 128;           // first row of block tile
    const long rbase     = blockBase + wave * 16;            // this wave's 16 rows

    // ---- stage all 1024 code norms (256 threads x 4 floats)
    *(f32x4*)&sC2[tid * 4] = *(const f32x4*)&c2[tid * 4];

    // ---- stage the 128-row input tile as bf16 into padded LDS (convert once)
    #pragma unroll 4
    for (int it = 0; it < 64; ++it) {
        int gi  = (it * 256 + tid) * 4;              // coalesced 16B chunks
        int row = gi >> 9, col = gi & 511;
        f32x4 f = *(const f32x4*)(x + (blockBase + row) * DD + col);
        v4bf o;
        o[0] = (__bf16)f[0]; o[1] = (__bf16)f[1];
        o[2] = (__bf16)f[2]; o[3] = (__bf16)f[3];
        *(v4bf*)(sX + row * LROW + col * 2) = o;
    }

    const unsigned char* bRow = sX + (wave * 16 + l16) * LROW + half * 16;
    const unsigned char* aRow = sCB + l16 * LROW + half * 16;

    // fragment loader: B = this wave's 16 rows, A0..A3 = 4x16 codes, D-chunk c
    auto ld = [&](ABu& B, ABu& A0, ABu& A1, ABu& A2, ABu& A3, int c) {
        const unsigned char* bp = bRow + c * 64;
        B.q[0] = *(const u32x4*)bp;
        B.q[1] = *(const u32x4*)(bp + 32);
        const unsigned char* ap = aRow + c * 64;
        A0.q[0] = *(const u32x4*)(ap + 0 * 16 * LROW);
        A0.q[1] = *(const u32x4*)(ap + 0 * 16 * LROW + 32);
        A1.q[0] = *(const u32x4*)(ap + 1 * 16 * LROW);
        A1.q[1] = *(const u32x4*)(ap + 1 * 16 * LROW + 32);
        A2.q[0] = *(const u32x4*)(ap + 2 * 16 * LROW);
        A2.q[1] = *(const u32x4*)(ap + 2 * 16 * LROW + 32);
        A3.q[0] = *(const u32x4*)(ap + 3 * 16 * LROW);
        A3.q[1] = *(const u32x4*)(ap + 3 * 16 * LROW + 32);
    };

    float best    = 3.4028235e38f;
    int   bestIdx = 0;

    // ---- scan codes in groups of 64 (LDS-staged), 4 WMMA accumulator tiles
    for (int g = 0; g < 16; ++g) {
        const int kbase = g * 64;
        __syncthreads();                         // prior group's LDS reads done

        // stage 64 bf16 codebook rows (1024B each); prefetch next tile
#if USE_ASYNC_LDS
        #pragma unroll
        for (int s = 0; s < 16; ++s) {
            int qi  = tid + s * 256;             // 0..4095 16B chunks
            int row = qi >> 6, col = qi & 63;
            const unsigned char* src =
                (const unsigned char*)(cb16 + (long)(kbase + row) * DD) + col * 16;
            __builtin_amdgcn_global_load_async_to_lds_b128(
                (AS1 v4i_vs*)src,
                (AS3 v4i_vs*)(sCB + row * LROW + col * 16), 0, 0);
            if (g < 15)
                __builtin_prefetch(src + 64 * DD * 2, 0, 1);
        }
        __builtin_amdgcn_s_wait_asynccnt(0);
#else
        {
            u32x4 tmp[16];
            #pragma unroll
            for (int s = 0; s < 16; ++s) {       // issue all 16 loads first
                int qi  = tid + s * 256;
                int row = qi >> 6, col = qi & 63;
                const unsigned char* src =
                    (const unsigned char*)(cb16 + (long)(kbase + row) * DD) + col * 16;
                tmp[s] = *(const u32x4*)src;
                if (g < 15)
                    __builtin_prefetch(src + 64 * DD * 2, 0, 1);
            }
            #pragma unroll
            for (int s = 0; s < 16; ++s) {       // then drain to LDS
                int qi  = tid + s * 256;
                int row = qi >> 6, col = qi & 63;
                *(u32x4*)(sCB + row * LROW + col * 16) = tmp[s];
            }
        }
#endif
        __syncthreads();

        v8f acc[4] = { (v8f){}, (v8f){}, (v8f){}, (v8f){} };

        // explicit ping-pong double buffering over the 16 D-chunks:
        // chunk c+1's 10 ds_loads are in flight while chunk c's WMMAs run.
        ABu Bp, Ap0, Ap1, Ap2, Ap3;
        ABu Bq, Aq0, Aq1, Aq2, Aq3;
        ld(Bp, Ap0, Ap1, Ap2, Ap3, 0);
        #pragma unroll
        for (int cc = 0; cc < 8; ++cc) {
            ld(Bq, Aq0, Aq1, Aq2, Aq3, 2 * cc + 1);
            acc[0] = __builtin_amdgcn_wmma_f32_16x16x32_bf16(
                         false, Ap0.v, false, Bp.v, (short)0, acc[0], false, false);
            acc[1] = __builtin_amdgcn_wmma_f32_16x16x32_bf16(
                         false, Ap1.v, false, Bp.v, (short)0, acc[1], false, false);
            acc[2] = __builtin_amdgcn_wmma_f32_16x16x32_bf16(
                         false, Ap2.v, false, Bp.v, (short)0, acc[2], false, false);
            acc[3] = __builtin_amdgcn_wmma_f32_16x16x32_bf16(
                         false, Ap3.v, false, Bp.v, (short)0, acc[3], false, false);
            if (cc < 7) ld(Bp, Ap0, Ap1, Ap2, Ap3, 2 * cc + 2);
            acc[0] = __builtin_amdgcn_wmma_f32_16x16x32_bf16(
                         false, Aq0.v, false, Bq.v, (short)0, acc[0], false, false);
            acc[1] = __builtin_amdgcn_wmma_f32_16x16x32_bf16(
                         false, Aq1.v, false, Bq.v, (short)0, acc[1], false, false);
            acc[2] = __builtin_amdgcn_wmma_f32_16x16x32_bf16(
                         false, Aq2.v, false, Bq.v, (short)0, acc[2], false, false);
            acc[3] = __builtin_amdgcn_wmma_f32_16x16x32_bf16(
                         false, Aq3.v, false, Bq.v, (short)0, acc[3], false, false);
        }

        // C layout: VGPR r, lane l -> code = r + 8*(l>>4), row = l&15.
        // score = ||c||^2 - 2*dot  (||x||^2 constant per row, dropped)
        #pragma unroll
        for (int t = 0; t < 4; ++t) {
            #pragma unroll
            for (int r = 0; r < 8; ++r) {
                int code = kbase + t * 16 + half * 8 + r;
                float score = sC2[code] - 2.0f * acc[t][r];
                if (score < best) { best = score; bestIdx = code; }
            }
        }
    }

    // combine the two wave halves (lane l and l^16 hold the same row)
    {
        float ov = __shfl_xor(best, 16, 32);
        int   oi = __shfl_xor(bestIdx, 16, 32);
        if (ov < best || (ov == best && oi < bestIdx)) { best = ov; bestIdx = oi; }
    }

    // ---- gather quantized rows (fp32, exact), loss partial, histogram
    float lsum = 0.0f;
    float* qout = out + 1;                        // d_out[0] = loss scalar
    #pragma unroll 1
    for (int j = 0; j < 16; ++j) {
        int  idx = __shfl(bestIdx, j, 32);        // row rbase+j's code
        long row = rbase + j;
        const float* crow = cb + (long)idx * DD + lane * 16;
        const float* xr   = x  + row * DD        + lane * 16;
        float*       orw  = qout + row * DD      + lane * 16;  // 4B-aligned only
        #pragma unroll
        for (int u = 0; u < 4; ++u) {
            f32x4 qv = *(const f32x4*)(crow + u * 4);
            f32x4 xv = *(const f32x4*)(xr   + u * 4);
            f32x4 dv = qv - xv;
            lsum += dv[0]*dv[0] + dv[1]*dv[1] + dv[2]*dv[2] + dv[3]*dv[3];
            orw[u*4+0] = qv[0]; orw[u*4+1] = qv[1];
            orw[u*4+2] = qv[2]; orw[u*4+3] = qv[3];
        }
        if (lane == j) atomicAdd(&hist[idx], 1u);
    }
    for (int off = 16; off; off >>= 1) lsum += __shfl_xor(lsum, off, 32);
    if (lane == 0) atomicAdd(sse, lsum);
}

// =====================================================================
// Kernel 3: finalize — loss scalar + perplexity.  1 block x 256.
// =====================================================================
__global__ __launch_bounds__(256)
void vq_fin(const unsigned* __restrict__ hist, const float* __restrict__ sse,
            float* __restrict__ out)
{
    __shared__ float red[256];
    const int t = threadIdx.x;
    float s = 0.0f;
    for (int k = t; k < KK; k += 256) {
        float p = (float)hist[k] * (1.0f / (float)NN);
        s += p * logf(p + 1e-10f);               // p==0 contributes 0
    }
    red[t] = s;
    __syncthreads();
    for (int off = 128; off; off >>= 1) {
        if (t < off) red[t] += red[t + off];
        __syncthreads();
    }
    if (t == 0) {
        // q_latent_loss + 0.25 * e_latent_loss, both == MSE forward
        out[0]         = 1.25f * sse[0] / (float)NDTOT;
        out[1 + NDTOT] = expf(-red[0]);          // perplexity
    }
}

// =====================================================================
extern "C" void kernel_launch(void* const* d_in, const int* in_sizes, int n_in,
                              void* d_out, int out_size, void* d_ws, size_t ws_size,
                              hipStream_t stream)
{
    (void)in_sizes; (void)n_in; (void)out_size; (void)ws_size;
    const float* x  = (const float*)d_in[0];     // inputs  [64,1024,512] f32
    const float* cb = (const float*)d_in[1];     // codebook [1024,512]  f32
    float* out = (float*)d_out;                  // [loss | quantized | perplexity]

    // workspace layout: sse(64B) | c2 4KB | hist 4KB | cb16 1MB
    float*          wsF  = (float*)d_ws;
    float*          sse  = wsF;                          // 1 float (padded 16)
    float*          c2   = wsF + 16;                     // KK floats
    unsigned*       hist = (unsigned*)(wsF + 16 + KK);   // KK uints
    unsigned short* cb16 = (unsigned short*)(wsF + 16 + 2 * KK);  // KK*DD bf16

    vq_prep<<<KK, 128, 0, stream>>>(cb, c2, cb16, hist, sse);
    vq_main<<<NN / 128, 256, SMEM_BYTES, stream>>>(x, cb, cb16, c2, hist, sse, out);
    vq_fin <<<1, 256, 0, stream>>>(hist, sse, out);
}
